// ASTGCN_52716428591149
// MI455X (gfx1250) — matrix operand order repaired and checked by reading, not compiled, not run on hardware
//
#include <hip/hip_runtime.h>
#include <hip/hip_bf16.h>

// ---------------- problem constants ----------------
constexpr int kB = 8, kN = 1536, kT = 12, kF = 64, kK = 3, kOut = 12;
constexpr size_t kNN = (size_t)kN * kN;
constexpr int kCT = kF * kT; // 768
constexpr float kLnEps = 1e-5f;

// Use GLOBAL_LOAD_ASYNC_TO_LDS_B128 + s_wait_asynccnt in the GEMM (gfx1250 path).
#define GEMM_ASYNC_LDS 1

typedef __attribute__((ext_vector_type(16))) __bf16 v16bf;
typedef __attribute__((ext_vector_type(8)))  float  v8f;
typedef __attribute__((ext_vector_type(2)))  float  v2f;

__device__ __forceinline__ float fast_rcp(float x) {
#if defined(__HIP_DEVICE_COMPILE__) && __has_builtin(__builtin_amdgcn_rcpf)
  return __builtin_amdgcn_rcpf(x);
#else
  return 1.0f / x;
#endif
}
__device__ __forceinline__ float sigmoidf_(float v) { return fast_rcp(1.0f + __expf(-v)); }

__device__ __forceinline__ float blockSum256(float v, float* red) {
  int tid = threadIdx.x;
  red[tid] = v; __syncthreads();
  for (int o = 128; o > 0; o >>= 1) { if (tid < o) red[tid] += red[tid + o]; __syncthreads(); }
  float r = red[0]; __syncthreads();
  return r;
}
__device__ __forceinline__ float blockMax256(float v, float* red) {
  int tid = threadIdx.x;
  red[tid] = v; __syncthreads();
  for (int o = 128; o > 0; o >>= 1) { if (tid < o) red[tid] = fmaxf(red[tid], red[tid + o]); __syncthreads(); }
  float r = red[0]; __syncthreads();
  return r;
}

// ---------------- temporal attention ----------------
// y[b,c,t] = sum_n x[b,n,c,t] * U1[n]
template<int C>
__global__ __launch_bounds__(256) void k_reduce_n(const float* __restrict__ x,
                                                  const float* __restrict__ U1,
                                                  float* __restrict__ y) {
  __shared__ float red[256];
  int t = blockIdx.x % kT;
  int c = (blockIdx.x / kT) % C;
  int b = blockIdx.x / (kT * C);
  float s = 0.f;
  for (int n = threadIdx.x; n < kN; n += 256)
    s += x[((size_t)(b * kN + n) * C + c) * kT + t] * U1[n];
  s = blockSum256(s, red);
  if (threadIdx.x == 0) y[(b * C + c) * kT + t] = s;
}

// lhs_t[b,t,m] = sum_c y[b,c,t] * U2[c,m]   (layout (B,T,N))
template<int C>
__global__ __launch_bounds__(256) void k_lhs_t(const float* __restrict__ y,
                                               const float* __restrict__ U2,
                                               float* __restrict__ lhs_t) {
  int idx = blockIdx.x * 256 + threadIdx.x;
  if (idx >= kB * kT * kN) return;
  int m = idx % kN;
  int t = (idx / kN) % kT;
  int b = idx / (kN * kT);
  float s = 0.f;
#pragma unroll
  for (int c = 0; c < C; ++c) s += y[(b * C + c) * kT + t] * U2[c * kN + m];
  lhs_t[idx] = s;
}

// rhs_t[b,n,t] = sum_c x[b,n,c,t] * U3[c]   (layout (B,N,T))
template<int C>
__global__ __launch_bounds__(256) void k_rhs_t(const float* __restrict__ x,
                                               const float* __restrict__ U3,
                                               float* __restrict__ rhs_t) {
  int idx = blockIdx.x * 256 + threadIdx.x;
  if (idx >= kB * kN * kT) return;
  int t = idx % kT;
  size_t bn = (size_t)(idx / kT);
  float s = 0.f;
#pragma unroll
  for (int c = 0; c < C; ++c) s += x[(bn * C + c) * kT + t] * U3[c];
  rhs_t[idx] = s;
}

// prod_t[b,t,s] = sum_n lhs_t[b,t,n]*rhs_t[b,n,s]
__global__ __launch_bounds__(256) void k_prod_t(const float* __restrict__ lhs_t,
                                                const float* __restrict__ rhs_t,
                                                float* __restrict__ prod_t) {
  __shared__ float red[256];
  int s_ = blockIdx.x % kT;
  int t = (blockIdx.x / kT) % kT;
  int b = blockIdx.x / (kT * kT);
  float a = 0.f;
  for (int n = threadIdx.x; n < kN; n += 256)
    a += lhs_t[((size_t)b * kT + t) * kN + n] * rhs_t[((size_t)b * kN + n) * kT + s_];
  a = blockSum256(a, red);
  if (threadIdx.x == 0) prod_t[(b * kT + t) * kT + s_] = a;
}

// E = softmax_t( Ve @ sigmoid(prod_t + be) )
__global__ __launch_bounds__(256) void k_E(const float* __restrict__ prod_t,
                                           const float* __restrict__ be,
                                           const float* __restrict__ Ve,
                                           float* __restrict__ E) {
  int b = blockIdx.x, tid = threadIdx.x;
  __shared__ float sg[kT * kT], tm[kT * kT];
  if (tid < kT * kT) sg[tid] = sigmoidf_(prod_t[b * kT * kT + tid] + be[tid]);
  __syncthreads();
  if (tid < kT * kT) {
    int t = tid / kT, u = tid % kT;
    float s = 0.f;
    for (int q = 0; q < kT; ++q) s += Ve[t * kT + q] * sg[q * kT + u];
    tm[tid] = s;
  }
  __syncthreads();
  if (tid < kT) {
    int u = tid;
    float mx = -1e30f;
    for (int t = 0; t < kT; ++t) mx = fmaxf(mx, tm[t * kT + u]);
    float se = 0.f;
    for (int t = 0; t < kT; ++t) se += __expf(tm[t * kT + u] - mx);
    float inv = fast_rcp(se);
    for (int t = 0; t < kT; ++t)
      E[b * kT * kT + t * kT + u] = __expf(tm[t * kT + u] - mx) * inv;
  }
}

// x_tat[b,n,c,u] = sum_t x[b,n,c,t]*E[b,t,u]
template<int C>
__global__ __launch_bounds__(256) void k_xtat(const float* __restrict__ x,
                                              const float* __restrict__ E,
                                              float* __restrict__ xt) {
  int b = blockIdx.y;
  __shared__ float Es[kT * kT];
  if (threadIdx.x < kT * kT) Es[threadIdx.x] = E[b * kT * kT + threadIdx.x];
  __syncthreads();
  int idx = blockIdx.x * 256 + threadIdx.x;
  if (idx >= kN * C) return;
  const float* xr = x + ((size_t)b * kN * C + idx) * kT;
  float* o = xt + ((size_t)b * kN * C + idx) * kT;
  float v[kT];
#pragma unroll
  for (int t = 0; t < kT; ++t) v[t] = xr[t];
#pragma unroll
  for (int u = 0; u < kT; ++u) {
    float s = 0.f;
#pragma unroll
    for (int t = 0; t < kT; ++t) s += v[t] * Es[t * kT + u];
    o[u] = s;
  }
}

// ---------------- spatial attention ----------------
// lhs_s[b,n,s] = sum_{c,t} xt[b,n,c,t]*W1[t]*W2[c,s]
template<int C>
__global__ __launch_bounds__(256) void k_lhs_s(const float* __restrict__ xt,
                                               const float* __restrict__ W1,
                                               const float* __restrict__ W2,
                                               float* __restrict__ lhs_s) {
  int idx = blockIdx.x * 256 + threadIdx.x;
  if (idx >= kB * kN) return;
  const float* xr = xt + (size_t)idx * C * kT;
  float v[C];
#pragma unroll
  for (int c = 0; c < C; ++c) {
    float s = 0.f;
#pragma unroll
    for (int t = 0; t < kT; ++t) s += xr[c * kT + t] * W1[t];
    v[c] = s;
  }
#pragma unroll
  for (int s_ = 0; s_ < kT; ++s_) {
    float a = 0.f;
#pragma unroll
    for (int c = 0; c < C; ++c) a += v[c] * W2[c * kT + s_];
    lhs_s[(size_t)idx * kT + s_] = a;
  }
}

// rhs_s[b,t,m] = sum_c xt[b,m,c,t]*W3[c]   (layout (B,T,N))
template<int C>
__global__ __launch_bounds__(256) void k_rhs_s(const float* __restrict__ xt,
                                               const float* __restrict__ W3,
                                               float* __restrict__ rhs_s) {
  int idx = blockIdx.x * 256 + threadIdx.x;
  if (idx >= kB * kT * kN) return;
  int m = idx % kN;
  int t = (idx / kN) % kT;
  int b = idx / (kN * kT);
  float s = 0.f;
#pragma unroll
  for (int c = 0; c < C; ++c) s += xt[((size_t)(b * kN + m) * C + c) * kT + t] * W3[c];
  rhs_s[idx] = s;
}

// sig[b,n,m] = sigmoid( lhs_s(Nx12) @ rhs_s(12xN) + bs )  -> bf16
// fp32 WMMA 16x16x4, K=12 in three steps. One 16x16 tile per wave.
__global__ __launch_bounds__(256) void k_prodS(const float* __restrict__ lhs_s,
                                               const float* __restrict__ rhs_s,
                                               const float* __restrict__ bs,
                                               __hip_bfloat16* __restrict__ sig) {
  const int wave = threadIdx.x >> 5, lane = threadIdx.x & 31;
  const int m_l = lane & 15, h = lane >> 4;
  const int tile_m = blockIdx.y * 8 + wave;
  const int tile_n = blockIdx.x;
  const int b = blockIdx.z;
  const int col = tile_n * 16 + m_l;
#if defined(__HIP_DEVICE_COMPILE__) && __has_builtin(__builtin_amdgcn_wmma_f32_16x16x4_f32)
  const int rowA = tile_m * 16 + m_l;
  const float* lrow = lhs_s + ((size_t)b * kN + rowA) * kT;
  v8f acc = {0.f, 0.f, 0.f, 0.f, 0.f, 0.f, 0.f, 0.f};
#pragma unroll
  for (int s = 0; s < 3; ++s) {
    int k0 = 4 * s + 2 * h;
    v2f a, bb;
    a.x = lrow[k0];
    a.y = lrow[k0 + 1];
    bb.x = rhs_s[((size_t)b * kT + k0) * kN + col];
    bb.y = rhs_s[((size_t)b * kT + k0 + 1) * kN + col];
    acc = __builtin_amdgcn_wmma_f32_16x16x4_f32(false, a, false, bb, (short)0, acc,
                                                false, false);
  }
#pragma unroll
  for (int i = 0; i < 8; ++i) {
    int row = tile_m * 16 + h * 8 + i;
    float v = sigmoidf_(acc[i] + bs[(size_t)row * kN + col]);
    sig[(size_t)b * kNN + (size_t)row * kN + col] = __float2bfloat16(v);
  }
#else
  for (int i = 0; i < 8; ++i) {
    int row = tile_m * 16 + h * 8 + i;
    float s = 0.f;
    for (int t = 0; t < kT; ++t)
      s += lhs_s[((size_t)b * kN + row) * kT + t] * rhs_s[((size_t)b * kT + t) * kN + col];
    float v = sigmoidf_(s + bs[(size_t)row * kN + col]);
    sig[(size_t)b * kNN + (size_t)row * kN + col] = __float2bfloat16(v);
  }
#endif
}

__global__ __launch_bounds__(256) void k_cvt_bf16(const float* __restrict__ in,
                                                  __hip_bfloat16* __restrict__ out, int n) {
  int i = blockIdx.x * 256 + threadIdx.x;
  if (i < n) out[i] = __float2bfloat16(in[i]);
}

// ---------------- generic bf16 WMMA GEMM: C(MxNcols) = A(MxK) * B(KxNcols) ----------------
// 128x128 workgroup tile, 8 waves (2 row x 4 col), each wave 64x32 via 8 accumulators.
// Global->LDS staging via GLOBAL_LOAD_ASYNC_TO_LDS_B128 (ASYNCcnt) with double buffering.
// A fragments load as two contiguous b128 LDS reads; B fragments are 16-bit K-pair gathers.
__global__ __launch_bounds__(256) void k_gemm_bf16(const __hip_bfloat16* __restrict__ A,
                                                   const __hip_bfloat16* __restrict__ Bm,
                                                   float* __restrict__ Cm,
                                                   int M, int Ncols, int Kd,
                                                   long long strideA, long long strideB,
                                                   long long strideC, int relu) {
  __shared__ __align__(16) __hip_bfloat16 As[2][128 * 32];
  __shared__ __align__(16) __hip_bfloat16 Bs[2][32 * 128];
  const int bz = blockIdx.z;
  A += (size_t)bz * strideA;
  Bm += (size_t)bz * strideB;
  Cm += (size_t)bz * strideC;
  const int tid = threadIdx.x;
  const int wave = tid >> 5, lane = tid & 31;
  const int wm = wave & 1, wn = wave >> 1;  // 2x4 waves, wave tile 64(M) x 32(N)
  const int m_l = lane & 15, h = lane >> 4;
  const int row0 = blockIdx.y * 128, col0 = blockIdx.x * 128;

  v8f acc[4][2];
  const v8f vz = {0.f, 0.f, 0.f, 0.f, 0.f, 0.f, 0.f, 0.f};
#pragma unroll
  for (int r = 0; r < 4; ++r)
#pragma unroll
    for (int c = 0; c < 2; ++c) acc[r][c] = vz;

  // Stage one 128x32 A tile + 32x128 B tile (8KB each) into LDS buffer `buf`.
  auto stage_tile = [&](int buf, int k0) {
#pragma unroll
    for (int i = 0; i < 2; ++i) {  // A tile: 512 x 16B chunks, 2 per thread
      int ci = tid * 2 + i;
      int r = ci >> 2, cc = ci & 3;
      const __hip_bfloat16* gp = &A[(size_t)(row0 + r) * Kd + k0 + cc * 8];
      __hip_bfloat16* lp = &As[buf][r * 32 + cc * 8];
#if GEMM_ASYNC_LDS
      unsigned lds = (unsigned)(uintptr_t)lp;  // LDS aperture: addr[31:0] is the LDS offset
      asm volatile("global_load_async_to_lds_b128 %0, %1, off"
                   :: "v"(lds), "v"((unsigned long long)(uintptr_t)gp)
                   : "memory");
#else
      *(uint4*)lp = *(const uint4*)gp;
#endif
    }
#pragma unroll
    for (int i = 0; i < 2; ++i) {  // B tile
      int ci = tid * 2 + i;
      int r = ci >> 4, cc = ci & 15;
      const __hip_bfloat16* gp = &Bm[(size_t)(k0 + r) * Ncols + col0 + cc * 8];
      __hip_bfloat16* lp = &Bs[buf][r * 128 + cc * 8];
#if GEMM_ASYNC_LDS
      unsigned lds = (unsigned)(uintptr_t)lp;
      asm volatile("global_load_async_to_lds_b128 %0, %1, off"
                   :: "v"(lds), "v"((unsigned long long)(uintptr_t)gp)
                   : "memory");
#else
      *(uint4*)lp = *(const uint4*)gp;
#endif
    }
  };

  stage_tile(0, 0);
#if GEMM_ASYNC_LDS
  asm volatile("s_wait_asynccnt 0x0" ::: "memory");
#endif
  __syncthreads();

  int buf = 0;
  for (int k0 = 0; k0 < Kd; k0 += 32) {
    if (k0 + 32 < Kd) stage_tile(buf ^ 1, k0 + 32);  // prefetch next tile into other buffer

    const uint4* AsQ = (const uint4*)As[buf];           // 4 words = 8 bf16 per quad
    const unsigned short* BsU = (const unsigned short*)Bs[buf];
    union Frag { v16bf v; unsigned int u[8]; uint4 q[2]; };
    Frag afr[4];
#pragma unroll
    for (int r = 0; r < 4; ++r) {
      // A fragment (16x32): for lane-half h, VGPRs 0-3 hold K-pairs at words [4h..4h+3],
      // VGPRs 4-7 at words [8+4h..8+4h+3] of the 16-word row -> two b128 LDS reads.
      int rowA = wm * 64 + r * 16 + m_l;
      afr[r].q[0] = AsQ[rowA * 4 + h];        // words 4h .. 4h+3
      afr[r].q[1] = AsQ[rowA * 4 + 2 + h];    // words 8+4h .. 8+4h+3
    }
    Frag bfr[2];
#pragma unroll
    for (int cb = 0; cb < 2; ++cb) {  // B fragment: K rows striped across lane halves
      int col = wn * 32 + cb * 16 + m_l;
#pragma unroll
      for (int vv = 0; vv < 8; ++vv) {
        int kk = 8 * h + 2 * vv + (vv >= 4 ? 8 : 0);
        unsigned int lo = BsU[kk * 128 + col];
        unsigned int hi = BsU[(kk + 1) * 128 + col];
        bfr[cb].u[vv] = lo | (hi << 16);
      }
    }
#pragma unroll
    for (int r = 0; r < 4; ++r)
#pragma unroll
      for (int cb = 0; cb < 2; ++cb)
        acc[r][cb] = __builtin_amdgcn_wmma_f32_16x16x32_bf16(
            false, afr[r].v, false, bfr[cb].v, (short)0, acc[r][cb], false, false);

#if GEMM_ASYNC_LDS
    asm volatile("s_wait_asynccnt 0x0" ::: "memory");
#endif
    __syncthreads();
    buf ^= 1;
  }

#pragma unroll
  for (int r = 0; r < 4; ++r) {
    int rb = row0 + wm * 64 + r * 16 + h * 8;
#pragma unroll
    for (int cb = 0; cb < 2; ++cb) {
      int col = col0 + wn * 32 + cb * 16 + m_l;
#pragma unroll
      for (int i = 0; i < 8; ++i) {
        float v = acc[r][cb][i];
        if (relu) v = fmaxf(v, 0.f);
        Cm[(size_t)(rb + i) * Ncols + col] = v;
      }
    }
  }
}

// softmax over n (axis=1) of P[b,:,m], fused with AT build:
// AT[b, m, k*N + n] = bf16( cheb[k][n,m] * S[b,n,m] )
__global__ __launch_bounds__(256) void k_softmax_AT(const float* __restrict__ P,
                                                    const float* __restrict__ cheb,
                                                    __hip_bfloat16* __restrict__ AT) {
  __shared__ float red[256];
  const int m = blockIdx.x, b = blockIdx.y;
  const float* Pc = P + (size_t)b * kNN + m;
  float mx = -1e30f;
  for (int n = threadIdx.x; n < kN; n += 256) mx = fmaxf(mx, Pc[(size_t)n * kN]);
  mx = blockMax256(mx, red);
  float se = 0.f;
  for (int n = threadIdx.x; n < kN; n += 256) se += __expf(Pc[(size_t)n * kN] - mx);
  se = blockSum256(se, red);
  float inv = fast_rcp(se);
  __hip_bfloat16* atr = AT + (size_t)(b * kN + m) * (size_t)(kK * kN);
  for (int n = threadIdx.x; n < kN; n += 256) {
    float s = __expf(Pc[(size_t)n * kN] - mx) * inv;
#pragma unroll
    for (int k = 0; k < kK; ++k)
      atr[k * kN + n] = __float2bfloat16(cheb[(size_t)k * kNN + (size_t)n * kN + m] * s);
  }
}

// Z[b, k*N+n, f*T+t] = sum_c x[b,n,c,t]*Theta[k][c,f]   (bf16, GEMM B-operand)
template<int C>
__global__ __launch_bounds__(256) void k_Z(const float* __restrict__ x,
                                           const float* __restrict__ Theta,
                                           __hip_bfloat16* __restrict__ Z) {
  int idx = blockIdx.x * 256 + threadIdx.x;
  if (idx >= kB * kK * kN * kF) return;
  int f = idx & 63;
  int rest = idx >> 6;
  int n = rest % kN;
  int rest2 = rest / kN;
  int k = rest2 % kK;
  int b = rest2 / kK;
  float a[kT];
#pragma unroll
  for (int t = 0; t < kT; ++t) a[t] = 0.f;
  const float* xr = x + (size_t)(b * kN + n) * C * kT;
  for (int c = 0; c < C; ++c) {
    float th = Theta[(k * C + c) * kF + f];
#pragma unroll
    for (int t = 0; t < kT; ++t) a[t] += xr[c * kT + t] * th;
  }
  __hip_bfloat16* zr = Z + ((size_t)(b * kK + k) * kN + n) * kCT + f * kT;
#pragma unroll
  for (int t = 0; t < kT; ++t) zr[t] = __float2bfloat16(a[t]);
}

// time conv (1x3) + residual 1x1 + relu + LayerNorm(F) -> out (B,N,F,T)
template<int C>
__global__ __launch_bounds__(256) void k_tc_ln(const float* __restrict__ gcn,
                                               const float* __restrict__ xin,
                                               const float* __restrict__ tcW,
                                               const float* __restrict__ tcb,
                                               const float* __restrict__ rcW,
                                               const float* __restrict__ rcb,
                                               const float* __restrict__ lng,
                                               const float* __restrict__ lnb,
                                               float* __restrict__ out) {
  const int bn = blockIdx.x;
  __shared__ float g[kCT], hb[kCT], mu[kT], rs[kT];
  const int tid = threadIdx.x;
  for (int i = tid; i < kCT; i += 256) g[i] = gcn[(size_t)bn * kCT + i];
  __syncthreads();
  for (int w = tid; w < kCT; w += 256) {
    int fo = w & 63, t = w >> 6;
    float tc = tcb[fo];
    for (int fi = 0; fi < kF; ++fi) {
      const float* wr = &tcW[(fo * kF + fi) * 3];
      float g0 = (t > 0) ? g[fi * kT + t - 1] : 0.f;
      float g1 = g[fi * kT + t];
      float g2 = (t < kT - 1) ? g[fi * kT + t + 1] : 0.f;
      tc += g0 * wr[0] + g1 * wr[1] + g2 * wr[2];
    }
    float res = rcb[fo];
    const float* xr = xin + (size_t)bn * C * kT;
#pragma unroll
    for (int c = 0; c < C; ++c) res += xr[c * kT + t] * rcW[fo * C + c];
    hb[t * kF + fo] = fmaxf(res + tc, 0.f);
  }
  __syncthreads();
  if (tid < kT) {
    float s = 0.f;
    for (int fo = 0; fo < kF; ++fo) s += hb[tid * kF + fo];
    float m = s * (1.f / kF);
    float v = 0.f;
    for (int fo = 0; fo < kF; ++fo) { float d = hb[tid * kF + fo] - m; v += d * d; }
    mu[tid] = m;
    rs[tid] = rsqrtf(v * (1.f / kF) + kLnEps);
  }
  __syncthreads();
  for (int w = tid; w < kCT; w += 256) {
    int fo = w & 63, t = w >> 6;
    float val = (hb[t * kF + fo] - mu[t]) * rs[t] * lng[fo] + lnb[fo];
    out[(size_t)bn * kCT + fo * kT + t] = val;
  }
}

// out[b,n,o] = sum_{f,t} h[b,n,f,t]*final_W[o,t,f] + final_b[o]
__global__ __launch_bounds__(256) void k_final(const float* __restrict__ h,
                                               const float* __restrict__ W,
                                               const float* __restrict__ bia,
                                               float* __restrict__ out) {
  int idx = blockIdx.x * 256 + threadIdx.x;
  if (idx >= kB * kN * kOut) return;
  int o = idx % kOut;
  size_t bn = (size_t)(idx / kOut);
  const float* hr = h + bn * kCT;
  const float* wr = W + o * kT * kF;
  float s = bia[o];
  for (int t = 0; t < kT; ++t)
    for (int f = 0; f < kF; ++f)
      s += hr[f * kT + t] * wr[t * kF + f];
  out[idx] = s;
}

// ---------------- host orchestration ----------------
struct Bufs {
  float *y, *lhs_t, *rhs_t, *prod_t, *E, *lhs_s, *rhs_s, *xtat, *P, *gcn;
  __hip_bfloat16 *sig, *vsbf, *AT, *Z;
};

template<int C>
static void run_block(const float* xin, const float* const* p, const float* cheb,
                      float* outp, const Bufs& w, hipStream_t stream) {
  const float *U1 = p[0], *U2 = p[1], *U3 = p[2], *be = p[3], *Ve = p[4],
              *W1 = p[5], *W2 = p[6], *W3 = p[7], *bs = p[8], *Vs = p[9],
              *Th = p[10], *tcW = p[11], *tcb = p[12], *rcW = p[13], *rcb = p[14],
              *lng = p[15], *lnb = p[16];
  k_reduce_n<C><<<dim3(kB * C * kT), 256, 0, stream>>>(xin, U1, w.y);
  k_lhs_t<C><<<dim3((kB * kT * kN + 255) / 256), 256, 0, stream>>>(w.y, U2, w.lhs_t);
  k_rhs_t<C><<<dim3((kB * kN * kT + 255) / 256), 256, 0, stream>>>(xin, U3, w.rhs_t);
  k_prod_t<<<dim3(kB * kT * kT), 256, 0, stream>>>(w.lhs_t, w.rhs_t, w.prod_t);
  k_E<<<dim3(kB), 256, 0, stream>>>(w.prod_t, be, Ve, w.E);
  k_xtat<C><<<dim3((kN * C + 255) / 256, kB), 256, 0, stream>>>(xin, w.E, w.xtat);
  k_lhs_s<C><<<dim3((kB * kN + 255) / 256), 256, 0, stream>>>(w.xtat, W1, W2, w.lhs_s);
  k_rhs_s<C><<<dim3((kB * kT * kN + 255) / 256), 256, 0, stream>>>(w.xtat, W3, w.rhs_s);
  k_prodS<<<dim3(kN / 16, kN / 16 / 8, kB), 256, 0, stream>>>(w.lhs_s, w.rhs_s, bs, w.sig);
  k_cvt_bf16<<<dim3((int)((kNN + 255) / 256)), 256, 0, stream>>>(Vs, w.vsbf, (int)kNN);
  // P[b] = Vs(bf16, shared) @ sig[b]    M=N, Ncols=N, K=N
  k_gemm_bf16<<<dim3(kN / 128, kN / 128, kB), 256, 0, stream>>>(
      w.vsbf, w.sig, w.P, kN, kN, kN, 0LL, (long long)kNN, (long long)kNN, 0);
  k_softmax_AT<<<dim3(kN, kB), 256, 0, stream>>>(w.P, cheb, w.AT);
  k_Z<C><<<dim3(kB * kK * kN * kF / 256), 256, 0, stream>>>(xin, Th, w.Z);
  // gcn[b] = relu( AT[b](Nx3N) @ Z[b](3Nx768) )
  k_gemm_bf16<<<dim3(kCT / 128, kN / 128, kB), 256, 0, stream>>>(
      w.AT, w.Z, w.gcn, kN, kCT, kK * kN,
      (long long)kN * kK * kN, (long long)kK * kN * kCT, (long long)kN * kCT, 1);
  k_tc_ln<C><<<dim3(kB * kN), 256, 0, stream>>>(w.gcn, xin, tcW, tcb, rcW, rcb, lng, lnb,
                                                outp);
}

extern "C" void kernel_launch(void* const* d_in, const int* in_sizes, int n_in,
                              void* d_out, int out_size, void* d_ws, size_t ws_size,
                              hipStream_t stream) {
  (void)in_sizes; (void)n_in; (void)out_size; (void)ws_size;
  const float* x = (const float*)d_in[0];
  const float* cheb = (const float*)d_in[1];
  const float* p0[17];
  const float* p1[17];
  for (int i = 0; i < 17; ++i) {
    p0[i] = (const float*)d_in[2 + i];
    p1[i] = (const float*)d_in[19 + i];
  }
  const float* finW = (const float*)d_in[36];
  const float* finB = (const float*)d_in[37];

  char* ws = (char*)d_ws;
  size_t off = 0;
  auto alloc = [&](size_t bytes) -> void* {
    off = (off + 255) & ~(size_t)255;
    void* p = ws + off;
    off += bytes;
    return p;
  };

  float* h1 = (float*)alloc((size_t)kB * kN * kCT * 4);
  float* h2 = (float*)alloc((size_t)kB * kN * kCT * 4);

  Bufs w;
  w.xtat = (float*)alloc((size_t)kB * kN * kF * kT * 4);
  w.gcn = w.xtat;  // alias: xtat is dead before gcn is produced (stream-ordered)
  w.sig = (__hip_bfloat16*)alloc((size_t)kB * kNN * 2);
  w.vsbf = (__hip_bfloat16*)alloc(kNN * 2);
  w.P = (float*)alloc((size_t)kB * kNN * 4);
  w.AT = (__hip_bfloat16*)alloc((size_t)kB * kN * (size_t)kK * kN * 2);
  w.Z = (__hip_bfloat16*)alloc((size_t)kB * kK * kN * (size_t)kCT * 2);
  w.y = (float*)alloc((size_t)kB * kF * kT * 4);
  w.lhs_t = (float*)alloc((size_t)kB * kT * kN * 4);
  w.rhs_t = (float*)alloc((size_t)kB * kN * kT * 4);
  w.prod_t = (float*)alloc((size_t)kB * kT * kT * 4);
  w.E = (float*)alloc((size_t)kB * kT * kT * 4);
  w.lhs_s = (float*)alloc((size_t)kB * kN * kT * 4);
  w.rhs_s = (float*)alloc((size_t)kB * kT * kN * 4);

  run_block<3>(x, p0, cheb, h1, w, stream);
  run_block<64>(h1, p1, cheb, h2, w, stream);
  k_final<<<dim3((kB * kN * kOut + 255) / 256), 256, 0, stream>>>(h2, finW, finB,
                                                                  (float*)d_out);
}
